// TFFastSpeechLengthRegulator_22591527977324
// MI455X (gfx1250) — 8-bit matrix operand order repaired and checked by reading, compiled, not run on hardware
//
#include <hip/hip_runtime.h>

#ifndef __has_builtin
#define __has_builtin(x) 0
#endif

typedef __attribute__((ext_vector_type(8))) int v8i;
typedef int v4i_ __attribute__((vector_size(16)));

#define BB 32
#define LL 256
#define HH 512

__global__ __launch_bounds__(32)
void cumsum_wmma_kernel(const int* __restrict__ dur,
                        int* __restrict__ csum,
                        int* __restrict__ total) {
    const int b    = blockIdx.x;
    const int lane = threadIdx.x;
    const int m    = lane & 15;
    const int hi   = lane >> 4;

    const int* drow = dur + b * LL;

    const int base = m * 16 + hi * 8;
    int a0 = (drow[base + 0] & 0xFF)
           | ((drow[base + 1] & 0xFF) << 8)
           | ((drow[base + 2] & 0xFF) << 16)
           | ((drow[base + 3] & 0xFF) << 24);
    int a1 = (drow[base + 4] & 0xFF)
           | ((drow[base + 5] & 0xFF) << 8)
           | ((drow[base + 6] & 0xFF) << 16)
           | ((drow[base + 7] & 0xFF) << 24);
    v8i A = {a0, a1, 0, 0, 0, 0, 0, 0};

    int b0 = 0, b1 = 0, b2 = 0, b3 = 0;
    {
        const int n = m;
        #pragma unroll
        for (int j = 0; j < 4; ++j) {
            if (hi * 8 + j     <= n) b0 |= 1 << (8 * j);
            if (hi * 8 + j + 4 <= n) b1 |= 1 << (8 * j);
        }
    }
    v8i Bm = {b0, b1, b2, b3, 0, 0, 0, 0};

    v8i C = {0, 0, 0, 0, 0, 0, 0, 0};
    v8i D = __builtin_amdgcn_wmma_i32_16x16x64_iu8(false, A, false, Bm, C,
                                                   false, false);

    int S[16];
    #pragma unroll
    for (int r = 0; r < 8; ++r) {
        S[r]     = __shfl(D[r], 15, 32);
        S[r + 8] = __shfl(D[r], 31, 32);
    }
    int O[16];
    int run = 0;
    #pragma unroll
    for (int i = 0; i < 16; ++i) { O[i] = run; run += S[i]; }

    #pragma unroll
    for (int r = 0; r < 8; ++r) {
        const int row = r + hi * 8;
        csum[b * LL + row * 16 + m] = D[r] + O[row];
    }
    if (lane == 0) total[b] = run;
}

__global__ __launch_bounds__(128)
void gather_kernel(const float* __restrict__ x,
                   const int* __restrict__ csum,
                   const int* __restrict__ total,
                   float* __restrict__ out,
                   float* __restrict__ mask,
                   int T) {
    __shared__ __align__(16) float rowbuf[HH];
    __shared__ int cs[LL];

    const int b   = blockIdx.y;
    const int t   = blockIdx.x;
    const int tid = threadIdx.x;

    float4* orow = (float4*)(out + ((size_t)b * T + t) * HH);

    const int tot = total[b];
    if (t >= tot) {
        float4 z = {0.f, 0.f, 0.f, 0.f};
        orow[tid] = z;
        if (tid == 0) mask[(size_t)b * T + t] = 0.f;
        return;
    }

    cs[tid]       = csum[b * LL + tid];
    cs[tid + 128] = csum[b * LL + tid + 128];
    __syncthreads();

    int lo = 0, hi = LL;
    while (lo < hi) {
        int mid = (lo + hi) >> 1;
        if (cs[mid] <= t) lo = mid + 1; else hi = mid;
    }
    const int idx = (lo < LL - 1) ? lo : (LL - 1);

    const float4* xrow = (const float4*)(x + ((size_t)b * LL + idx) * HH);

#if __has_builtin(__builtin_amdgcn_global_load_async_to_lds_b128) && \
    __has_builtin(__builtin_amdgcn_global_store_async_from_lds_b128)
    typedef __attribute__((address_space(1))) v4i_ gv4_t;
    typedef __attribute__((address_space(3))) v4i_ lv4_t;
    gv4_t* gsrc = (gv4_t*)(xrow + tid);
    lv4_t* lptr = (lv4_t*)(rowbuf + tid * 4);
    gv4_t* gdst = (gv4_t*)(orow + tid);
    __builtin_amdgcn_global_load_async_to_lds_b128(gsrc, lptr, 0, 0);
#if __has_builtin(__builtin_amdgcn_s_wait_asynccnt)
    __builtin_amdgcn_s_wait_asynccnt(0);
#else
    asm volatile("s_wait_asynccnt 0" ::: "memory");
#endif
    __builtin_amdgcn_global_store_async_from_lds_b128(gdst, lptr, 0, 0);
#else
    orow[tid] = xrow[tid];
#endif

    if (tid == 0) mask[(size_t)b * T + t] = 1.f;
}

extern "C" void kernel_launch(void* const* d_in, const int* in_sizes, int n_in,
                              void* d_out, int out_size, void* d_ws, size_t ws_size,
                              hipStream_t stream) {
    const float* x   = (const float*)d_in[0];
    const int*   dur = (const int*)d_in[1];

    const int T = out_size / (BB * (HH + 1));

    int* csum  = (int*)d_ws;
    int* total = csum + BB * LL;

    float* out  = (float*)d_out;
    float* mask = out + (size_t)BB * T * HH;

    cumsum_wmma_kernel<<<BB, 32, 0, stream>>>(dur, csum, total);

    dim3 grid((unsigned)T, BB);
    gather_kernel<<<grid, 128, 0, stream>>>(x, csum, total, out, mask, T);
}
